// SeparableWeightedConv1D_23785528886122
// MI455X (gfx1250) — matrix operand order repaired and checked by reading, not compiled, not run on hardware
//
#include <hip/hip_runtime.h>
#include <math.h>

typedef __bf16 bf16_t;
typedef bf16_t v8bf  __attribute__((ext_vector_type(8)));
typedef bf16_t v16bf __attribute__((ext_vector_type(16)));
typedef float  v8f   __attribute__((ext_vector_type(8)));

#define B_ 4
#define T_ 4
#define C_ 32
#define N_ 16384
#define F_ 64
#define K_ 9
#define PAD_ 4
#define CK_ (C_ * K_)        // 288
#define NTILE 32             // n-columns per block
#define LST 296              // LDS row stride in bf16 elements (mult of 8 -> 16B rows; 148 dwords % 64 = 20 -> conflict-free)

// f32 -> bf16 RNE split helpers (bit-exact, no NaN inputs here)
__device__ __forceinline__ unsigned short f2bf(float f) {
    unsigned u = __float_as_uint(f);
    unsigned r = u + 0x7FFFu + ((u >> 16) & 1u);
    return (unsigned short)(r >> 16);
}
__device__ __forceinline__ float bf2f(unsigned short h) {
    return __uint_as_float(((unsigned)h) << 16);
}
__device__ __forceinline__ v16bf cat8(v8bf a, v8bf b) {
    return __builtin_shufflevector(a, b, 0,1,2,3,4,5,6,7,8,9,10,11,12,13,14,15);
}

// Per (b,t): out[f,n] = sum_ck A[f,ck] * Bmat[ck,n]  via split-bf16 WMMA:
//   A ~ Ah + Al, B ~ Bh + Bl;  acc += Ah*Bh + Ah*Bl + Al*Bh   (f32 accumulate)
__global__ __launch_bounds__(256)
void sepconv_wmma_bf16x3(const float* __restrict__ x,
                         const float* __restrict__ coords,
                         const float* __restrict__ weight,
                         const int*   __restrict__ sigma_p,
                         float* __restrict__ out) {
    extern __shared__ __align__(16) unsigned char smem_raw[];
    unsigned short* Ah = (unsigned short*)smem_raw;                 // [F_][LST]
    unsigned short* Al = Ah + F_ * LST;                             // [F_][LST]
    unsigned short* Bh = Al + F_ * LST;                             // [NTILE][LST] (transposed: n-major, ck inner)
    unsigned short* Bl = Bh + NTILE * LST;                          // [NTILE][LST]
    float*          Wk = (float*)(Bl + NTILE * LST);                // [K_ * NTILE] distance weights

    const int tid = threadIdx.x;
    const int bt  = blockIdx.y;                // 0..B*T-1
    const int t   = bt & (T_ - 1);
    const int n0  = blockIdx.x * NTILE;
    const float inv_sigma = 1.0f / (float)sigma_p[0];

    // ---- Stage A = weight[t] : 64 x 288 contiguous floats, split hi/lo ----
    const float* wsrc = weight + (size_t)t * F_ * CK_;
    for (int i = tid; i < F_ * CK_; i += 256) {
        const float w = wsrc[i];
        const unsigned short h = f2bf(w);
        const unsigned short l = f2bf(w - bf2f(h));
        const int idx = (i / CK_) * LST + (i % CK_);
        Ah[idx] = h;
        Al[idx] = l;
    }

    // ---- Distance weights w[k][n] for this n-tile ----
    const float* cb = coords + (size_t)bt * 3 * N_;
    for (int i = tid; i < K_ * NTILE; i += 256) {
        const int k  = i >> 5;                 // 0..8
        const int n  = i & 31;
        const int nc = n0 + n;                 // center tap (in range)
        const int ns = nc + k - PAD_;          // window tap
        float w = 0.0f;                        // OOB: x_pad==0 -> product 0, weight irrelevant
        if (ns >= 0 && ns < N_) {
            const float dx = cb[ns]          - cb[nc];
            const float dy = cb[N_ + ns]     - cb[N_ + nc];
            const float dz = cb[2 * N_ + ns] - cb[2 * N_ + nc];
            const float d2 = dx * dx + dy * dy + dz * dz;
            const float d  = (d2 > 0.0f) ? sqrtf(d2) : 0.0f;
            w = fmaxf(1.0f - d * inv_sigma, 0.0f);
        }
        Wk[i] = w;
    }
    __syncthreads();

    // ---- Stage Bmat transposed + split: Bs[n][ck] = x_win * distw ----
    const float* xb = x + (size_t)bt * C_ * N_;
    for (int i = tid; i < CK_ * NTILE; i += 256) {
        const int ck = i >> 5;                 // 0..287
        const int n  = i & 31;
        const int c  = ck / K_;
        const int k  = ck - c * K_;
        const int ns = n0 + n + k - PAD_;
        const float xv = (ns >= 0 && ns < N_) ? xb[c * N_ + ns] : 0.0f;
        const float v  = xv * Wk[k * NTILE + n];
        const unsigned short h = f2bf(v);
        const unsigned short l = f2bf(v - bf2f(h));
        const int idx = n * LST + ck;
        Bh[idx] = h;
        Bl[idx] = l;
    }
    __syncthreads();

    // ---- WMMA: each wave computes one 16(f) x 16(n) tile; 9 chunks x 3 wmma ----
    const int wave  = tid >> 5;                // 0..7
    const int lane  = tid & 31;
    const int lo    = lane & 15;
    const int hi    = lane >> 4;
    const int ftile = wave & 3;                // 4 f-tiles cover F_=64
    const int ntile = wave >> 2;               // 2 n-tiles cover NTILE=32

    // A 16x32 bf16 layout: lane holds K = 8*hi..+7 (v0-3) and 16+8*hi..+7 (v4-7) of row lo
    const unsigned short* ArH = Ah + (ftile * 16 + lo) * LST;
    const unsigned short* ArL = Al + (ftile * 16 + lo) * LST;
    // B 32x16 bf16 layout: lane holds K = 8*hi..+7 and 16+8*hi..+7 at col lo (matches A)
    const unsigned short* BrH = Bh + (ntile * 16 + lo) * LST;
    const unsigned short* BrL = Bl + (ntile * 16 + lo) * LST;
    const int a0 = 8 * hi;                     // first A half-fragment offset
    const int a1 = 16 + 8 * hi;                // second A half-fragment offset
    const int b0 = 8 * hi;                     // B fragment offset

    v8f acc = {};
    #pragma unroll
    for (int ck0 = 0; ck0 < CK_; ck0 += 32) {
        const v16bf ah = cat8(*(const v8bf*)(ArH + ck0 + a0), *(const v8bf*)(ArH + ck0 + a1));
        const v16bf al = cat8(*(const v8bf*)(ArL + ck0 + a0), *(const v8bf*)(ArL + ck0 + a1));
        const v16bf bh = cat8(*(const v8bf*)(BrH + ck0 + b0), *(const v8bf*)(BrH + ck0 + b0 + 16));
        const v16bf bl = cat8(*(const v8bf*)(BrL + ck0 + b0), *(const v8bf*)(BrL + ck0 + b0 + 16));
        acc = __builtin_amdgcn_wmma_f32_16x16x32_bf16(false, ah, false, bh, (short)0, acc, false, false);
        acc = __builtin_amdgcn_wmma_f32_16x16x32_bf16(false, ah, false, bl, (short)0, acc, false, false);
        acc = __builtin_amdgcn_wmma_f32_16x16x32_bf16(false, al, false, bh, (short)0, acc, false, false);
    }

    // ---- Store D: VGPR r, lanes 0-15: M=r; lanes 16-31: M=r+8; N=lane%16 ----
    float* op = out + ((size_t)bt * F_ + ftile * 16) * N_ + (n0 + ntile * 16 + lo);
    #pragma unroll
    for (int r = 0; r < 8; ++r) {
        op[(size_t)(r + 8 * hi) * N_] = acc[r];
    }
}

extern "C" void kernel_launch(void* const* d_in, const int* in_sizes, int n_in,
                              void* d_out, int out_size, void* d_ws, size_t ws_size,
                              hipStream_t stream) {
    const float* x      = (const float*)d_in[0];
    const float* coords = (const float*)d_in[1];
    const float* weight = (const float*)d_in[2];
    const int*   sigma  = (const int*)d_in[3];
    float* out = (float*)d_out;

    (void)in_sizes; (void)n_in; (void)out_size; (void)d_ws; (void)ws_size;

    dim3 grid(N_ / NTILE, B_ * T_);            // 512 x 16 blocks
    dim3 block(256);                           // 8 wave32s
    const size_t shmem = (size_t)(2 * F_ * LST + 2 * NTILE * LST) * sizeof(unsigned short)
                       + (size_t)(K_ * NTILE) * sizeof(float);     // ~115 KB
    hipLaunchKernelGGL(sepconv_wmma_bf16x3, grid, block, shmem, stream,
                       x, coords, weight, sigma, out);
}
